// PureElmanCell_2396591751750
// MI455X (gfx1250) — compile-verified, matrix operand order checked by reading, not run on hardware
//
#include <hip/hip_runtime.h>
#include <math.h>

// Problem constants (reference): T=2048, B=32, D=1024
#define T_STEPS 2048
#define BATCH   32
#define DIM     1024
#define M_ROWS  (T_STEPS * BATCH)                  // 65536
#define HS_ELEMS ((size_t)T_STEPS * BATCH * DIM)   // hs; h_all region follows in d_out

#define NWG 16                      // phase-2 cooperative workgroups (one WGP each)
#define COLS_PER_WG (DIM / NWG)     // 64
#define WPITCH 1032                 // padded LDS pitch (ushorts): 2064B rows, 16B aligned
#define HPITCH 1032

typedef __attribute__((ext_vector_type(16))) __bf16 v16bf;
typedef __attribute__((ext_vector_type(8)))  float  v8f;

union Frag16 { v16bf v; uint4 q[2]; };

__device__ __forceinline__ unsigned short f2bf(float x) {
    unsigned u = __builtin_bit_cast(unsigned, x);
    return (unsigned short)((u + 0x7FFFu + ((u >> 16) & 1u)) >> 16);  // RNE
}

// gfx1250 async copy: 16B global -> LDS, tracked by ASYNCcnt
__device__ __forceinline__ void async_b128(const void* lds, const void* g) {
    asm volatile("global_load_async_to_lds_b128 %0, %1, off"
                 :: "v"((unsigned)(size_t)lds),
                    "v"((unsigned long long)(size_t)g) : "memory");
}

// ---------------------------------------------------------------------------
// Pass 0a: x (fp32) -> bf16, element-wise. 4 elems/thread, float4 in, ushort4 out.
// ---------------------------------------------------------------------------
__global__ __launch_bounds__(256) void cvt_x(
    const float* __restrict__ x, unsigned short* __restrict__ xbf)
{
    const size_t i = ((size_t)blockIdx.x * 256 + threadIdx.x) * 4;
    const float4 f = *(const float4*)(x + i);
    ushort4 o;
    o.x = f2bf(f.x); o.y = f2bf(f.y); o.z = f2bf(f.z); o.w = f2bf(f.w);
    *(ushort4*)(xbf + i) = o;
}

// ---------------------------------------------------------------------------
// Pass 0b: transpose + convert a DIMxDIM fp32 matrix: outT[n][k] = bf16(in[k][n]).
// 64x64 tiles via LDS, coalesced on both sides.
// ---------------------------------------------------------------------------
__global__ __launch_bounds__(256) void cvt_transpose(
    const float* __restrict__ in, unsigned short* __restrict__ outT)
{
    __shared__ unsigned short t[64][72];        // pad kills bank conflicts
    const int k0 = (blockIdx.x & 15) * 64;
    const int n0 = (blockIdx.x >> 4) * 64;
    const int tid = threadIdx.x;
    #pragma unroll
    for (int j = 0; j < 16; ++j) {
        const int e = tid + j * 256, r = e >> 6, c = e & 63;   // r:k, c:n
        t[c][r] = f2bf(in[(size_t)(k0 + r) * DIM + (n0 + c)]);
    }
    __syncthreads();
    #pragma unroll
    for (int j = 0; j < 16; ++j) {
        const int e = tid + j * 256, r = e >> 6, c = e & 63;   // r:n, c:k
        outT[(size_t)(n0 + r) * DIM + (k0 + c)] = t[r][c];
    }
}

// ---------------------------------------------------------------------------
// Init (runs AFTER phase-1 GEMM): zero barrier counter, h_all[0]=h0, hbuf0=bf16(h0)
// ---------------------------------------------------------------------------
__global__ __launch_bounds__(256) void elman_init(
    const float* __restrict__ h0, float* __restrict__ out,
    unsigned short* __restrict__ hbuf0, unsigned* __restrict__ cnt)
{
    const int i = blockIdx.x * 256 + threadIdx.x;
    if (i == 0) *cnt = 0u;
    if (i < BATCH * DIM) {
        const float v = h0[i];
        out[HS_ELEMS + i] = v;   // h_all[0]
        hbuf0[i] = f2bf(v);
    }
}

// ---------------------------------------------------------------------------
// Phase 1: xW = x @ W_x + b, all-bf16 inputs, async double-buffered LDS staging.
// Block = 256 threads (8 waves), tile 128(M) x 64(N), K-chunks of 32.
// Pipeline: issue async copy of chunk k+1, s_wait_asynccnt 3 (retires chunk k),
// one barrier, 4 WMMAs/wave, barrier.
// ---------------------------------------------------------------------------
__global__ __launch_bounds__(256) void elman_xw_gemm(
    const unsigned short* __restrict__ xbf,   // [M][DIM] bf16
    const unsigned short* __restrict__ WxT,   // [DIM n][DIM k] bf16 (pre-transposed)
    const float* __restrict__ b,
    float* __restrict__ xw_out)
{
    __shared__ alignas(16) unsigned short Ab[2][128][40];  // 80B rows: 16B aligned
    __shared__ alignas(16) unsigned short Bb[2][64][40];

    const int tid  = threadIdx.x;
    const int lane = tid & 31;
    const int wave = tid >> 5;
    const int c0 = blockIdx.x * 64;    // N offset
    const int r0 = blockIdx.y * 128;   // M offset

    // Stage one 32-wide K chunk into LDS buffer `buf`: 3 async b128 per thread
    auto stage = [&](int buf, int kk) {
        {   const int c = tid;                 // A chunks 0..255
            async_b128(&Ab[buf][c >> 2][(c & 3) * 8],
                       xbf + (((size_t)(r0 + (c >> 2))) << 10) + kk + (c & 3) * 8);
        }
        {   const int c = tid + 256;           // A chunks 256..511
            async_b128(&Ab[buf][c >> 2][(c & 3) * 8],
                       xbf + (((size_t)(r0 + (c >> 2))) << 10) + kk + (c & 3) * 8);
        }
        {   const int c = tid;                 // B chunks 0..255
            async_b128(&Bb[buf][c >> 2][(c & 3) * 8],
                       WxT + (((size_t)(c0 + (c >> 2))) << 10) + kk + (c & 3) * 8);
        }
    };

    stage(0, 0);
    v8f acc[4] = {};
    const int khalf = (lane >> 4) * 8;

    #pragma unroll 2
    for (int kk = 0; kk < DIM; kk += 32) {
        const int buf = (kk >> 5) & 1;
        if (kk + 32 < DIM) {
            stage(buf ^ 1, kk + 32);
            asm volatile("s_wait_asynccnt 0x3" ::: "memory");  // chunk kk landed
        } else {
            asm volatile("s_wait_asynccnt 0x0" ::: "memory");
        }
        __syncthreads();

        Frag16 a;
        const unsigned short* ar = &Ab[buf][wave * 16 + (lane & 15)][khalf];
        a.q[0] = *(const uint4*)ar;
        a.q[1] = *(const uint4*)(ar + 16);
        #pragma unroll
        for (int j = 0; j < 4; ++j) {
            Frag16 bfr;
            const unsigned short* br = &Bb[buf][j * 16 + (lane & 15)][khalf];
            bfr.q[0] = *(const uint4*)br;
            bfr.q[1] = *(const uint4*)(br + 16);
            acc[j] = __builtin_amdgcn_wmma_f32_16x16x32_bf16(
                false, a.v, false, bfr.v, (short)0, acc[j], false, false);
        }
        __syncthreads();
    }

    // Epilogue: C layout — VGPR r: lanes0-15 M=r, lanes16-31 M=r+8; N=lane%16
    #pragma unroll
    for (int j = 0; j < 4; ++j) {
        const int col = c0 + j * 16 + (lane & 15);
        const float bias = b[col];
        #pragma unroll
        for (int r = 0; r < 8; ++r) {
            const int m = r0 + wave * 16 + ((lane >> 4) << 3) + r;
            xw_out[(size_t)m * DIM + col] = acc[j][r] + bias;
        }
    }
}

// ---------------------------------------------------------------------------
// Phase 2: persistent cooperative recurrence.
// 16 WGs x 256 threads; WG g owns columns [g*64, g*64+64).
// W_hT slice (64x1024 bf16 = 128KB) async-staged into LDS ONCE for all steps.
// Per step: async-broadcast h_{t-1} (64KB) L2->LDS, 32 WMMAs/wave, tanh,
// write hs/h_all + bf16 ping-pong buffer, one monotonic grid barrier.
// ---------------------------------------------------------------------------
__global__ __launch_bounds__(256) void elman_recurrence(
    const unsigned short* __restrict__ WhT,   // [DIM n][DIM k] bf16 (pre-transposed)
    float* __restrict__ out,
    unsigned short* __restrict__ hbuf0, unsigned short* __restrict__ hbuf1,
    unsigned* __restrict__ barrier_cnt)
{
    extern __shared__ char smem_raw[];
    unsigned short* Wl  = (unsigned short*)smem_raw;                             // [64][WPITCH]
    unsigned short* hsm = (unsigned short*)(smem_raw + (size_t)64 * WPITCH * 2); // [32][HPITCH]

    const int tid  = threadIdx.x;
    const int lane = tid & 31;
    const int wave = tid >> 5;
    const int n0   = blockIdx.x * COLS_PER_WG;

    // One-time: async-stage the W_hT column slice (64 rows x 1024 k = 8192 x 16B)
    #pragma unroll 4
    for (int j = 0; j < 32; ++j) {
        const int c = j * 256 + tid;               // 16B-chunk index
        const int n = c >> 7, cc = c & 127;
        async_b128(&Wl[n * WPITCH + cc * 8],
                   WhT + (size_t)(n0 + n) * DIM + cc * 8);
    }
    asm volatile("s_wait_asynccnt 0x0" ::: "memory");
    __syncthreads();

    const int mt = wave >> 2;          // M tile: rows 0-15 / 16-31
    const int nt = wave & 3;           // N tile within the 64-col slice
    const int khalf = (lane >> 4) * 8;

    for (int t = 0; t < T_STEPS; ++t) {
        const unsigned short* hprev = (t & 1) ? hbuf1 : hbuf0;
        unsigned short*       hnext = (t & 1) ? hbuf0 : hbuf1;

        // Async copy h_{t-1} (32x1024 bf16 = 4096 x 16B) into LDS
        #pragma unroll 4
        for (int it = 0; it < 16; ++it) {
            const int idx = it * 256 + tid;
            async_b128(&hsm[(idx >> 7) * HPITCH + (idx & 127) * 8],
                       hprev + idx * 8);
        }
        asm volatile("s_wait_asynccnt 0x0" ::: "memory");
        __syncthreads();

        // acc = h_{t-1} @ W_h (slice): 32 K-chunks, one WMMA each
        v8f acc = {};
        #pragma unroll 8
        for (int kk = 0; kk < DIM; kk += 32) {
            Frag16 a, bfr;
            const unsigned short* ar =
                &hsm[(mt * 16 + (lane & 15)) * HPITCH + kk + khalf];
            a.q[0] = *(const uint4*)ar;
            a.q[1] = *(const uint4*)(ar + 16);
            const unsigned short* br =
                &Wl[(nt * 16 + (lane & 15)) * WPITCH + kk + khalf];
            bfr.q[0] = *(const uint4*)br;
            bfr.q[1] = *(const uint4*)(br + 16);
            acc = __builtin_amdgcn_wmma_f32_16x16x32_bf16(
                false, a.v, false, bfr.v, (short)0, acc, false, false);
        }

        // h_t = tanh(acc + xW_t); xW was staged in hs[t], overwritten in place
        const int n = n0 + nt * 16 + (lane & 15);
        #pragma unroll
        for (int r = 0; r < 8; ++r) {
            const int m = mt * 16 + ((lane >> 4) << 3) + r;
            const size_t idx = ((size_t)t * BATCH + m) * DIM + n;
            const float h = tanhf(acc[r] + out[idx]);
            out[idx] = h;                                           // hs[t]
            out[HS_ELEMS + idx + (size_t)BATCH * DIM] = h;          // h_all[t+1]
            hnext[m * DIM + n] = f2bf(h);
        }

        // Grid barrier: monotonic counter, one atomic per WG
        __syncthreads();
        if (tid == 0) {
            __threadfence();
            atomicAdd(barrier_cnt, 1u);
            const unsigned target = (unsigned)(t + 1) * NWG;
            while (__hip_atomic_load(barrier_cnt, __ATOMIC_ACQUIRE,
                                     __HIP_MEMORY_SCOPE_AGENT) < target)
                __builtin_amdgcn_s_sleep(1);
        }
        __syncthreads();
    }
}

// ---------------------------------------------------------------------------
extern "C" void kernel_launch(void* const* d_in, const int* in_sizes, int n_in,
                              void* d_out, int out_size, void* d_ws, size_t ws_size,
                              hipStream_t stream)
{
    const float* x  = (const float*)d_in[0];   // [T,B,D]
    const float* h0 = (const float*)d_in[1];   // [B,D]
    const float* Wx = (const float*)d_in[2];   // [D,D]
    const float* Wh = (const float*)d_in[3];   // [D,D]
    const float* b  = (const float*)d_in[4];   // [D]
    float* out = (float*)d_out;                // hs [T,B,D] ++ h_all [T+1,B,D]

    // bf16 staging lives inside the h_all region (dead until phase 2 writes it):
    //   xbf:  M*D ushorts (134MB), WxT/WhT: D*D ushorts (2MB each). Total 138MB < 268MB.
    unsigned short* xbf = (unsigned short*)(out + HS_ELEMS);
    unsigned short* WxT = xbf + (size_t)M_ROWS * DIM;
    unsigned short* WhT = WxT + (size_t)DIM * DIM;

    char* ws = (char*)d_ws;
    unsigned*       cnt   = (unsigned*)ws;                                  // barrier
    unsigned short* hbuf0 = (unsigned short*)(ws + 256);                    // 64KB
    unsigned short* hbuf1 = (unsigned short*)(ws + 256 + BATCH * DIM * 2);  // 64KB

    // Pass 0: one-time bf16 conversion / transposition (memory-bound, ~17us)
    cvt_x<<<(M_ROWS * (size_t)DIM) / 1024, 256, 0, stream>>>(x, xbf);
    cvt_transpose<<<256, 256, 0, stream>>>(Wx, WxT);
    cvt_transpose<<<256, 256, 0, stream>>>(Wh, WhT);

    // Phase 1: big GEMM, xW -> hs region
    dim3 g1(DIM / 64, M_ROWS / 128);
    elman_xw_gemm<<<g1, 256, 0, stream>>>(xbf, WxT, b, out);

    // Init AFTER gemm (h_all[0] overlaps the xbf staging bytes)
    elman_init<<<(BATCH * DIM) / 256, 256, 0, stream>>>(h0, out, hbuf0, cnt);

    // Phase 2: cooperative recurrence, ~194KB LDS per WG
    const size_t smem = ((size_t)64 * WPITCH + (size_t)32 * HPITCH) * 2;
    elman_recurrence<<<NWG, 256, smem, stream>>>(WhT, out, hbuf0, hbuf1, cnt);
}
